// T2MUnet_86638080295507
// MI455X (gfx1250) — matrix operand order repaired and checked by reading, not compiled, not run on hardware
//
#include <hip/hip_runtime.h>
#include <hip/hip_bf16.h>

typedef __attribute__((ext_vector_type(16))) _Float16 v16h;
typedef __attribute__((ext_vector_type(8)))  float    v8f;

#define TILE_M 64
#define TILE_N 64
#define TILE_K 32
#define AT_LD  40   // padded LDS row stride (halves): 80B, 16B-aligned, breaks bank periodicity
#define BT_LD  40

// ---------------- math helpers ----------------
__device__ __forceinline__ float mishf(float x) {
  float sp = (x > 20.f) ? x : log1pf(expf(x));
  return x * tanhf(sp);
}

// ---------------- WMMA fragment helpers (wave32, contiguous k = 16*(lane/16)+s in both A and B) ----------------
// A tile in LDS: row-major [TILE_M][AT_LD] f16.
// A 16x32 frag: lane<16 -> row=lane,    K = {0..15}
//               lane>=16 -> row=lane-16, K = {16..31}
__device__ __forceinline__ v16h load_a_frag(const _Float16* At, int m0) {
  int lane = threadIdx.x & 31;
  int row  = m0 + (lane & 15);
  int k0   = (lane < 16) ? 0 : 16;
  const _Float16* p = At + row * AT_LD;
  v16h f;
#pragma unroll
  for (int j = 0; j < 8; ++j) { f[j] = p[k0 + j]; f[8 + j] = p[k0 + 8 + j]; }
  return f;
}

// B tile in LDS: n-major [TILE_N][BT_LD] f16 (transposed K x N tile).
// B 32x16 frag: half j holds K=(lane<16 ? j : 16+j), N = lane&15
__device__ __forceinline__ v16h load_b_frag(const _Float16* Bt, int n0) {
  int lane = threadIdx.x & 31;
  int n    = n0 + (lane & 15);
  int k0   = (lane < 16) ? 0 : 16;
  const _Float16* p = Bt + n * BT_LD + k0;
  v16h f;
#pragma unroll
  for (int j = 0; j < 16; ++j) f[j] = p[j];
  return f;
}

__device__ __forceinline__ v8f wmma32(v16h a, v16h b, v8f c) {
  return __builtin_amdgcn_wmma_f32_16x16x32_f16(false, a, false, b, (short)0, c,
                                                false, false);
}

// ---------------- generic batched WMMA GEMM: C[m][n] = sum_k A[m][k]*B[k][n] ----------------
// A: [M][lda] f16 (strideA per z, 0 => shared weights)
// B: [K][ldb] f16 (strideB per z)
// C: [M][ldc] f32 (strideC per z); bias per-row (optional); accumulate: C += result
// nguard: 1 => N not a multiple of TILE_N; B loads are still unconditional (rows
// over-read into adjacent scratch), out-of-range elements are zeroed by selects.
__launch_bounds__(128)
__global__ void wmma_gemm(const _Float16* __restrict__ A, int lda, long long strideA,
                          const _Float16* __restrict__ B, int ldb, long long strideB,
                          float* __restrict__ C, int ldc, long long strideC,
                          const float* __restrict__ bias,
                          int M, int N, int K, int accumulate, int nguard) {
  __shared__ _Float16 At[TILE_M * AT_LD];
  __shared__ _Float16 Bt[TILE_N * BT_LD];

  int z  = blockIdx.z;
  const _Float16* Ab = A + (size_t)z * (size_t)strideA;
  const _Float16* Bb = B + (size_t)z * (size_t)strideB;
  float*          Cb = C + (size_t)z * (size_t)strideC;

  int m0 = blockIdx.y * TILE_M;
  int n0 = blockIdx.x * TILE_N;
  int t  = threadIdx.x;
  int hsel = (t & 1) * 16;   // A-copy: which 16-half K chunk
  int rowA = t >> 1;         // A-copy: tile row 0..63
  int krow = t >> 2;         // B-copy: K row 0..31
  int nch  = (t & 3) * 16;   // B-copy: N chunk 0/16/32/48

  v8f acc[2][2];
#pragma unroll
  for (int i = 0; i < 2; ++i)
#pragma unroll
    for (int j = 0; j < 2; ++j) acc[i][j] = v8f{};

  int w  = t >> 5;
  int wm = (w >> 1) * 32;
  int wn = (w & 1) * 32;

  for (int k0 = 0; k0 < K; k0 += TILE_K) {
    // A tile: contiguous 32B chunks, row-major
    {
      const _Float16* pa = Ab + (size_t)(m0 + rowA) * lda + k0 + hsel;
      _Float16* da = At + rowA * AT_LD + hsel;
#pragma unroll
      for (int j = 0; j < 16; ++j) da[j] = pa[j];
    }
    // B tile: unconditional contiguous 32B read of one K-row x 16 N-cols,
    // transpose via LDS scatter -> Bt[n][k]
    {
      const _Float16* pb = Bb + (size_t)(k0 + krow) * ldb + n0 + nch;
      _Float16 tmp[16];
#pragma unroll
      for (int j = 0; j < 16; ++j) tmp[j] = pb[j];
      if (nguard) {   // wave-uniform scalar branch; only K/V projections take it
#pragma unroll
        for (int j = 0; j < 16; ++j)
          if (n0 + nch + j >= N) tmp[j] = (_Float16)0.f;
      }
#pragma unroll
      for (int j = 0; j < 16; ++j) Bt[(nch + j) * BT_LD + krow] = tmp[j];
    }
    __syncthreads();

    v16h a0 = load_a_frag(At, wm);
    v16h a1 = load_a_frag(At, wm + 16);
    v16h b0 = load_b_frag(Bt, wn);
    v16h b1 = load_b_frag(Bt, wn + 16);
    acc[0][0] = wmma32(a0, b0, acc[0][0]);
    acc[0][1] = wmma32(a0, b1, acc[0][1]);
    acc[1][0] = wmma32(a1, b0, acc[1][0]);
    acc[1][1] = wmma32(a1, b1, acc[1][1]);
    __syncthreads();
  }

  // epilogue: C/D layout — VGPR r: lane<16 -> M=r, lane>=16 -> M=8+r; N=lane&15
  int lane = t & 31;
  int ncl  = lane & 15;
  int mb   = (lane < 16) ? 0 : 8;
#pragma unroll
  for (int i = 0; i < 2; ++i)
#pragma unroll
    for (int j = 0; j < 2; ++j) {
      int gn = n0 + wn + j * 16 + ncl;
      if (gn >= N) continue;
#pragma unroll
      for (int r = 0; r < 8; ++r) {
        int gm = m0 + wm + i * 16 + mb + r;
        size_t idx = (size_t)gm * ldc + gn;
        float v = acc[i][j][r];
        if (bias) v += bias[gm];
        if (accumulate) v += Cb[idx];
        Cb[idx] = v;
      }
    }
}

// ---------------- implicit-im2col WMMA conv (5 taps, pad 2) ----------------
// W: [M][Cin*5] f16 (kk = ci*5+tap, matches (O,I,H) flat layout)
// X: per-batch [Cin][L+4] f16 with zeroed 2-halo (branch-free im2col)
// C: per-batch [M][L] f32
__launch_bounds__(128)
__global__ void wmma_conv5(const _Float16* __restrict__ W,
                           const _Float16* __restrict__ X, long long strideX,
                           float* __restrict__ C, long long strideC,
                           const float* __restrict__ bias,
                           int M, int L, int Cin) {
  __shared__ _Float16 At[TILE_M * AT_LD];
  __shared__ _Float16 Bt[TILE_N * BT_LD];

  int K = Cin * 5;
  int Lp = L + 4;
  int z = blockIdx.z;
  const _Float16* Xb = X + (size_t)z * (size_t)strideX;
  float*          Cb = C + (size_t)z * (size_t)strideC;

  int m0 = blockIdx.y * TILE_M;
  int n0 = blockIdx.x * TILE_N;
  int t  = threadIdx.x;
  int hsel = (t & 1) * 16;
  int rowA = t >> 1;
  int krow = t >> 2;
  int nch  = (t & 3) * 16;

  v8f acc[2][2];
#pragma unroll
  for (int i = 0; i < 2; ++i)
#pragma unroll
    for (int j = 0; j < 2; ++j) acc[i][j] = v8f{};

  int w  = t >> 5;
  int wm = (w >> 1) * 32;
  int wn = (w & 1) * 32;

  for (int k0 = 0; k0 < K; k0 += TILE_K) {
    {
      const _Float16* pa = W + (size_t)(m0 + rowA) * K + k0 + hsel;
      _Float16* da = At + rowA * AT_LD + hsel;
#pragma unroll
      for (int j = 0; j < 16; ++j) da[j] = pa[j];
    }
    // im2col row: fixed kk -> (ci,tap); 16 consecutive l are contiguous in padded X
    {
      int kk  = k0 + krow;
      int ci  = kk / 5;
      int tap = kk - ci * 5;
      const _Float16* pb = Xb + (size_t)ci * Lp + n0 + nch + tap;
      _Float16 tmp[16];
#pragma unroll
      for (int j = 0; j < 16; ++j) tmp[j] = pb[j];
#pragma unroll
      for (int j = 0; j < 16; ++j) Bt[(nch + j) * BT_LD + krow] = tmp[j];
    }
    __syncthreads();

    v16h a0 = load_a_frag(At, wm);
    v16h a1 = load_a_frag(At, wm + 16);
    v16h b0 = load_b_frag(Bt, wn);
    v16h b1 = load_b_frag(Bt, wn + 16);
    acc[0][0] = wmma32(a0, b0, acc[0][0]);
    acc[0][1] = wmma32(a0, b1, acc[0][1]);
    acc[1][0] = wmma32(a1, b0, acc[1][0]);
    acc[1][1] = wmma32(a1, b1, acc[1][1]);
    __syncthreads();
  }

  int lane = t & 31;
  int ncl  = lane & 15;
  int mb   = (lane < 16) ? 0 : 8;
#pragma unroll
  for (int i = 0; i < 2; ++i)
#pragma unroll
    for (int j = 0; j < 2; ++j) {
      int gn = n0 + wn + j * 16 + ncl;
#pragma unroll
      for (int r = 0; r < 8; ++r) {
        int gm = m0 + wm + i * 16 + mb + r;
        Cb[(size_t)gm * L + gn] = acc[i][j][r] + bias[gm];
      }
    }
}

// ---------------- elementwise / norm kernels ----------------
__global__ void zero_f16(_Float16* __restrict__ d, long long n) {
  long long i = (long long)blockIdx.x * blockDim.x + threadIdx.x;
  long long st = (long long)gridDim.x * blockDim.x;
  for (; i < n; i += st) d[i] = (_Float16)0.f;
}

__global__ void f32_to_f16(const float* __restrict__ s, _Float16* __restrict__ d,
                           long long n) {
  long long i = (long long)blockIdx.x * blockDim.x + threadIdx.x;
  long long st = (long long)gridDim.x * blockDim.x;
  for (; i < n; i += st) d[i] = (_Float16)s[i];
}

// convert f32 [rows][L] -> f16 padded [rows][L+4] at column offset 2
__global__ void f32_to_f16_pad(const float* __restrict__ s, _Float16* __restrict__ d,
                               long long rows, int L) {
  long long n = rows * L;
  long long i = (long long)blockIdx.x * blockDim.x + threadIdx.x;
  long long st = (long long)gridDim.x * blockDim.x;
  for (; i < n; i += st) {
    long long r = i / L;
    int l = (int)(i - r * L);
    d[r * (L + 4) + l + 2] = (_Float16)s[i];
  }
}

// ss[b][o] = sum_i mish(t[b][i]) * tm_w[o][i] + tm_b[o]   (64 x 1024, K=512)
__global__ void time_mlp(const float* __restrict__ t, const float* __restrict__ w,
                         const float* __restrict__ bias, float* __restrict__ ss) {
  int o = blockIdx.x * blockDim.x + threadIdx.x;  // 0 .. 64*1024-1
  int b = o >> 10, oo = o & 1023;
  const float* tp = t + (size_t)b * 512;
  const float* wp = w + (size_t)oo * 512;
  float acc = bias[oo];
  for (int i = 0; i < 512; ++i) acc += mishf(tp[i]) * wp[i];
  ss[o] = acc;
}

// per-(b,l): GroupNorm over 8 groups of 64 ch, FiLM with ss, mish, emit padded f16
__launch_bounds__(512)
__global__ void gn0_film_mish(const float* __restrict__ h, const float* __restrict__ ss,
                              const float* __restrict__ g, const float* __restrict__ bt,
                              _Float16* __restrict__ outh, int L) {
  int bl = blockIdx.x;
  int b = bl / L, l = bl - b * L;
  int c = threadIdx.x;  // 0..511
  float v = h[((size_t)b * 512 + c) * L + l];
  __shared__ float s1[512], s2[512];
  s1[c] = v; s2[c] = v * v;
  __syncthreads();
  for (int s = 32; s > 0; s >>= 1) {
    if ((c & 63) < s) { s1[c] += s1[c + s]; s2[c] += s2[c + s]; }
    __syncthreads();
  }
  int gb = c & ~63;
  float mean = s1[gb] * (1.f / 64.f);
  float var  = s2[gb] * (1.f / 64.f) - mean * mean;
  float xn = (v - mean) * rsqrtf(var + 1e-5f);
  float y = xn * g[c] + bt[c];
  float scale = ss[(size_t)b * 1024 + c];
  float shift = ss[(size_t)b * 1024 + 512 + c];
  y = mishf(y * (1.f + scale) + shift);
  outh[((size_t)b * 512 + c) * (L + 4) + l + 2] = (_Float16)y;
}

// per-(b,l): GN(8x64)+mish on h1, add res, write out(f32); then LayerNorm over
// all 512 channels and emit f16 operand for the Q GEMM (channel-major).
__launch_bounds__(512)
__global__ void gn1_res_out_ln(const float* __restrict__ h, const float* __restrict__ res,
                               const float* __restrict__ g1, const float* __restrict__ b1,
                               const float* __restrict__ lng, const float* __restrict__ lnb,
                               float* __restrict__ out, _Float16* __restrict__ xlnh, int L) {
  int bl = blockIdx.x;
  int b = bl / L, l = bl - b * L;
  int c = threadIdx.x;
  size_t idx = ((size_t)b * 512 + c) * L + l;
  float v = h[idx];
  __shared__ float s1[512], s2[512];
  s1[c] = v; s2[c] = v * v;
  __syncthreads();
  for (int s = 32; s > 0; s >>= 1) {
    if ((c & 63) < s) { s1[c] += s1[c + s]; s2[c] += s2[c + s]; }
    __syncthreads();
  }
  int gb = c & ~63;
  float mean = s1[gb] * (1.f / 64.f);
  float var  = s2[gb] * (1.f / 64.f) - mean * mean;
  float xn = (v - mean) * rsqrtf(var + 1e-5f);
  float y = mishf(xn * g1[c] + b1[c]) + res[idx];
  out[idx] = y;
  __syncthreads();
  // LayerNorm over 512 channels
  s1[c] = y; s2[c] = y * y;
  __syncthreads();
  for (int s = 256; s > 0; s >>= 1) {
    if (c < s) { s1[c] += s1[c + s]; s2[c] += s2[c + s]; }
    __syncthreads();
  }
  float m2 = s1[0] * (1.f / 512.f);
  float v2 = s2[0] * (1.f / 512.f) - m2 * m2;
  float x2 = (y - m2) * rsqrtf(v2 + 1e-5f);
  xlnh[idx] = (_Float16)(x2 * lng[c] + lnb[c]);
}

// LayerNorm(cond) over 768, emit f16 transposed to [b][768][80] (N padded 77->80)
__launch_bounds__(256)
__global__ void ln_cond(const float* __restrict__ cond, const float* __restrict__ g,
                        const float* __restrict__ bt, _Float16* __restrict__ cnh) {
  int bn = blockIdx.x;
  int b = bn / 77, n = bn - b * 77;
  int t = threadIdx.x;  // 256 threads, 3 cols each
  const float* src = cond + ((size_t)b * 77 + n) * 768;
  float v0 = src[t], v1 = src[t + 256], v2 = src[t + 512];
  __shared__ float s1[256], s2[256];
  s1[t] = v0 + v1 + v2;
  s2[t] = v0 * v0 + v1 * v1 + v2 * v2;
  __syncthreads();
  for (int s = 128; s > 0; s >>= 1) {
    if (t < s) { s1[t] += s1[t + s]; s2[t] += s2[t + s]; }
    __syncthreads();
  }
  float mean = s1[0] * (1.f / 768.f);
  float var  = s2[0] * (1.f / 768.f) - mean * mean;
  float rs = rsqrtf(var + 1e-5f);
  float vv[3] = {v0, v1, v2};
#pragma unroll
  for (int k = 0; k < 3; ++k) {
    int d = t + k * 256;
    cnh[((size_t)b * 768 + d) * 80 + n] = (_Float16)((vv[k] - mean) * rs * g[d] + bt[d]);
  }
}

// softmax over Dh=64 per (b, t, head): q stored [b][d][t]
__launch_bounds__(512)
__global__ void softmax_q(const float* __restrict__ q, _Float16* __restrict__ qsh) {
  int bt = blockIdx.x;
  int b = bt >> 9, t = bt & 511;
  int c = threadIdx.x;
  size_t idx = ((size_t)b * 512 + c) * 512 + t;
  float v = q[idx];
  __shared__ float s1[512];
  s1[c] = v;
  __syncthreads();
  for (int s = 32; s > 0; s >>= 1) {
    if ((c & 63) < s) s1[c] = fmaxf(s1[c], s1[c + s]);
    __syncthreads();
  }
  float mx = s1[c & ~63];
  __syncthreads();
  float e = expf(v - mx);
  s1[c] = e;
  __syncthreads();
  for (int s = 32; s > 0; s >>= 1) {
    if ((c & 63) < s) s1[c] += s1[c + s];
    __syncthreads();
  }
  float sum = s1[c & ~63];
  qsh[idx] = (_Float16)(e / sum);
}

// softmax over n (77) per (b, d), in place; k stored [b][512][80]
__global__ void softmax_k(float* __restrict__ k) {
  int i = blockIdx.x * blockDim.x + threadIdx.x;
  if (i >= 64 * 512) return;
  float* row = k + (size_t)i * 80;
  float mx = -1e30f;
  for (int n = 0; n < 77; ++n) mx = fmaxf(mx, row[n]);
  float s = 0.f;
  for (int n = 0; n < 77; ++n) s += expf(row[n] - mx);
  float inv = 1.f / s;
  for (int n = 0; n < 77; ++n) row[n] = expf(row[n] - mx) * inv;
}

// attnT[z][dl][dh] = sum_{n<77} ks[b][h*64+dh][n] * vs[b][h*64+dl][n], z=b*8+h
__global__ void attn_kernel(const float* __restrict__ ks, const float* __restrict__ vs,
                            _Float16* __restrict__ attnT) {
  int idx = blockIdx.x * blockDim.x + threadIdx.x;  // 512*4096
  int z = idx >> 12;
  int r = idx & 4095;
  int dl = r >> 6, dh = r & 63;
  int b = z >> 3, h = z & 7;
  const float* kr = ks + ((size_t)b * 512 + h * 64 + dh) * 80;
  const float* vr = vs + ((size_t)b * 512 + h * 64 + dl) * 80;
  float acc = 0.f;
  for (int n = 0; n < 77; ++n) acc += kr[n] * vr[n];
  attnT[(size_t)z * 4096 + dl * 64 + dh] = (_Float16)acc;
}

// ---------------- host launch ----------------
extern "C" void kernel_launch(void* const* d_in, const int* in_sizes, int n_in,
                              void* d_out, int out_size, void* d_ws, size_t ws_size,
                              hipStream_t stream) {
  (void)in_sizes; (void)n_in; (void)out_size; (void)ws_size;
  const int B = 64, CIN = 256, COUT = 512, L = 512, CD = 768;
  const int LP = L + 4;

  const float* x       = (const float*)d_in[0];
  const float* tt      = (const float*)d_in[1];
  const float* cond    = (const float*)d_in[2];
  const float* conv0_w = (const float*)d_in[3];
  const float* conv0_b = (const float*)d_in[4];
  const float* gn0_g   = (const float*)d_in[5];
  const float* gn0_b   = (const float*)d_in[6];
  const float* tm_w    = (const float*)d_in[7];
  const float* tm_b    = (const float*)d_in[8];
  const float* conv1_w = (const float*)d_in[9];
  const float* conv1_b = (const float*)d_in[10];
  const float* gn1_g   = (const float*)d_in[11];
  const float* gn1_b   = (const float*)d_in[12];
  const float* res_w   = (const float*)d_in[13];
  const float* res_b   = (const float*)d_in[14];
  const float* ln_x_g  = (const float*)d_in[15];
  const float* ln_x_b  = (const float*)d_in[16];
  const float* ln_c_g  = (const float*)d_in[17];
  const float* ln_c_b  = (const float*)d_in[18];
  const float* q_w     = (const float*)d_in[19];
  const float* q_b     = (const float*)d_in[20];
  const float* k_w     = (const float*)d_in[21];
  const float* k_b     = (const float*)d_in[22];
  const float* v_w     = (const float*)d_in[23];
  const float* v_b     = (const float*)d_in[24];
  // d_in[25] = cond_indices = arange(B): identity gather, folded away.
  float* out = (float*)d_out;

  // ---- workspace layout (bump allocator, 256B aligned) ----
  char* p = (char*)d_ws;
  auto alloc = [&](size_t bytes) -> char* {
    char* r = p;
    p += (bytes + 255) & ~(size_t)255;
    return r;
  };
  _Float16* xh    = (_Float16*)alloc((size_t)B * CIN * LP * 2);   // padded halo layout
  _Float16* w0h   = (_Float16*)alloc((size_t)COUT * CIN * 5 * 2);
  _Float16* w1h   = (_Float16*)alloc((size_t)COUT * COUT * 5 * 2);
  _Float16* wrh   = (_Float16*)alloc((size_t)COUT * CIN * 2);
  _Float16* qwh   = (_Float16*)alloc((size_t)COUT * COUT * 2);
  _Float16* kwh   = (_Float16*)alloc((size_t)COUT * CD * 2);
  _Float16* vwh   = (_Float16*)alloc((size_t)COUT * CD * 2);
  float*    ss    = (float*)alloc((size_t)B * 1024 * 4);
  float*    h0    = (float*)alloc((size_t)B * COUT * L * 4);      // conv0 out; reused as h1, then qf
  _Float16* h0mh  = (_Float16*)alloc((size_t)B * COUT * LP * 2);  // padded halo layout
  float*    resb  = (float*)alloc((size_t)B * COUT * L * 4);
  _Float16* xlnh  = (_Float16*)alloc((size_t)B * COUT * L * 2);
  float*    kf    = (float*)alloc((size_t)B * COUT * 80 * 4);
  float*    vf    = (float*)alloc((size_t)B * COUT * 80 * 4);
  _Float16* cnh   = (_Float16*)alloc((size_t)B * CD * 80 * 2);
  _Float16* attnT = (_Float16*)alloc((size_t)512 * 64 * 64 * 2);
  // disjoint live-range aliases
  float*    h1    = h0;               // conv1 output (h0 consumed by gn0 kernel)
  float*    qf    = h0;               // q projection (h1 consumed by gn1 kernel)
  _Float16* qsh   = (_Float16*)resb;  // res consumed by gn1 kernel before softmax_q

  // ---- zero halos, then f32 -> f16 conversions ----
  zero_f16<<<2048, 256, 0, stream>>>(xh, (long long)B * CIN * LP);
  zero_f16<<<2048, 256, 0, stream>>>(h0mh, (long long)B * COUT * LP);
  f32_to_f16_pad<<<2048, 256, 0, stream>>>(x, xh, (long long)B * CIN, L);
  f32_to_f16<<<512, 256, 0, stream>>>(conv0_w, w0h, (long long)COUT * CIN * 5);
  f32_to_f16<<<512, 256, 0, stream>>>(conv1_w, w1h, (long long)COUT * COUT * 5);
  f32_to_f16<<<512, 256, 0, stream>>>(res_w, wrh, (long long)COUT * CIN);
  f32_to_f16<<<512, 256, 0, stream>>>(q_w, qwh, (long long)COUT * COUT);
  f32_to_f16<<<512, 256, 0, stream>>>(k_w, kwh, (long long)COUT * CD);
  f32_to_f16<<<512, 256, 0, stream>>>(v_w, vwh, (long long)COUT * CD);

  // ---- time embedding MLP ----
  time_mlp<<<(B * 1024) / 256, 256, 0, stream>>>(tt, tm_w, tm_b, ss);

  // ---- conv0 (implicit-GEMM WMMA): M=512, N=L=512, K=256*5 ----
  wmma_conv5<<<dim3(L / 64, COUT / 64, B), 128, 0, stream>>>(
      w0h, xh, (long long)CIN * LP, h0, (long long)COUT * L, conv0_b, COUT, L, CIN);

  // ---- GN0 + FiLM + mish -> padded f16 ----
  gn0_film_mish<<<B * L, 512, 0, stream>>>(h0, ss, gn0_g, gn0_b, h0mh, L);

  // ---- conv1: M=512, N=512, K=512*5 ----
  wmma_conv5<<<dim3(L / 64, COUT / 64, B), 128, 0, stream>>>(
      w1h, h0mh, (long long)COUT * LP, h1, (long long)COUT * L, conv1_b, COUT, L, COUT);

  // ---- residual 1x1 conv as GEMM: M=512, N=512, K=256 (reads padded xh, base+2) ----
  wmma_gemm<<<dim3(L / 64, COUT / 64, B), 128, 0, stream>>>(
      wrh, CIN, 0, xh + 2, LP, (long long)CIN * LP, resb, L, (long long)COUT * L,
      res_b, COUT, L, CIN, 0, 0);

  // ---- GN1 + mish + residual add -> d_out; LayerNorm -> xlnh ----
  gn1_res_out_ln<<<B * L, 512, 0, stream>>>(h1, resb, gn1_g, gn1_b, ln_x_g, ln_x_b,
                                            out, xlnh, L);

  // ---- LayerNorm(cond) -> cnh [b][768][80] ----
  ln_cond<<<B * 77, 256, 0, stream>>>(cond, ln_c_g, ln_c_b, cnh);

  // ---- Q projection: M=512, N=512, K=512 ----
  wmma_gemm<<<dim3(L / 64, COUT / 64, B), 128, 0, stream>>>(
      qwh, COUT, 0, xlnh, L, (long long)COUT * L, qf, L, (long long)COUT * L,
      q_b, COUT, L, COUT, 0, 0);

  // ---- K/V projections: M=512, N=77(pad 80), K=768 ----
  wmma_gemm<<<dim3(2, COUT / 64, B), 128, 0, stream>>>(
      kwh, CD, 0, cnh, 80, (long long)CD * 80, kf, 80, (long long)COUT * 80,
      k_b, COUT, 77, CD, 0, 1);
  wmma_gemm<<<dim3(2, COUT / 64, B), 128, 0, stream>>>(
      vwh, CD, 0, cnh, 80, (long long)CD * 80, vf, 80, (long long)COUT * 80,
      v_b, COUT, 77, CD, 0, 1);

  // ---- softmaxes ----
  softmax_q<<<B * L, 512, 0, stream>>>(qf, qsh);
  softmax_k<<<(B * COUT) / 256, 256, 0, stream>>>(kf);

  // ---- attn = k^T v per (b,h): 64x64, K=77 (tiny, VALU) ----
  attn_kernel<<<(512 * 4096) / 256, 256, 0, stream>>>(kf, vf, attnT);

  // ---- y = attnT @ qs, accumulated into out: per z=(b,h) M=64,N=512,K=64 ----
  wmma_gemm<<<dim3(L / 64, 1, 512), 128, 0, stream>>>(
      attnT, 64, 64 * 64, qsh, L, (long long)64 * L, out, L, (long long)64 * L,
      nullptr, 64, L, 64, 1, 0);
}
